// DLRM_5437428597128
// MI455X (gfx1250) — matrix operand order repaired and checked by reading, not compiled, not run on hardware
//
#include <hip/hip_runtime.h>
#include <math.h>

typedef _Float16 half_t;
typedef __attribute__((ext_vector_type(16))) _Float16 v16h;
typedef __attribute__((ext_vector_type(8)))  _Float16 v8h;
typedef __attribute__((ext_vector_type(8)))  float    v8f;

#define BATCH   16384
#define COUNT   200000
#define NUM_CAT 26
#define ROWS    64          // rows per block (4 M-tiles of 16)
#define NWAVES  8           // 256 threads / wave32

// ---------------------------------------------------------------------------
// Pack f32 weights [K][N] into WMMA B-fragment layout, f16, zero-padded.
// Fragment (nT, kt) = 32 lanes x 16 halves, contiguous 1KB block:
//   dst[((nT*KT + kt)*32 + lane)*16 + j]  <-  W[kt*32 + 16*(j>>3) + 8*(lane>>4) + (j&7)][nT*16 + (lane&15)]
// ---------------------------------------------------------------------------
__global__ void pack_b_kernel(const float* __restrict__ src, half_t* __restrict__ dst,
                              int K, int N, int KT, int NT) {
    int i = blockIdx.x * blockDim.x + threadIdx.x;
    if (i >= KT * NT * 512) return;
    int frag = i >> 9, rem = i & 511;
    int lane = rem >> 4, j = rem & 15;
    int nT = frag / KT, kt = frag - nT * KT;
    int k = kt * 32 + ((j >> 3) << 4) + ((lane >> 4) << 3) + (j & 7);
    int n = (nT << 4) + (lane & 15);
    float v = (k < K && n < N) ? src[(size_t)k * N + n] : 0.0f;
    dst[i] = (half_t)v;
}

// ---------------------------------------------------------------------------
// One GEMM layer: act[ROWS][Kpad] (LDS f16) x packed-W + bias (+ReLU) -> LDS f16.
// Each wave-unit = one nTile x two mTiles; B fragment loaded once, used twice.
// ---------------------------------------------------------------------------
__device__ __forceinline__ void gemm_layer(const half_t* __restrict__ actIn, int ldA,
                                           const half_t* __restrict__ Wp, int KT,
                                           const float*  __restrict__ bias,
                                           half_t* __restrict__ actOut, int ldO,
                                           int nTiles, bool relu, int tid) {
    const int lane = tid & 31, wave = tid >> 5;
    const int lm = lane & 15, hi = lane >> 4;
    const int units = nTiles * 2;
    for (int u = wave; u < units; u += NWAVES) {
        const int nT    = u >> 1;
        const int mBase = (u & 1) << 5;                 // 0 or 32
        const int col   = (nT << 4) + lm;
        const half_t* bfrag = Wp + (((size_t)nT * KT) << 9) + (lane << 4);
        const half_t* ar0   = actIn + (size_t)(mBase + lm) * ldA + (hi << 3);
        const half_t* ar1   = ar0 + (size_t)16 * ldA;
        v8f c0 = {}, c1 = {};
        for (int kt = 0; kt < KT; ++kt) {
            v16h b = *(const v16h*)(bfrag + ((size_t)kt << 9));   // 2x b128, coalesced
            const half_t* p0 = ar0 + (kt << 5);
            const half_t* p1 = ar1 + (kt << 5);
            v8h l0 = *(const v8h*)p0, h0 = *(const v8h*)(p0 + 16);
            v8h l1 = *(const v8h*)p1, h1 = *(const v8h*)(p1 + 16);
            v16h a0, a1;
#pragma unroll
            for (int j = 0; j < 8; ++j) {
                a0[j] = l0[j]; a0[j + 8] = h0[j];
                a1[j] = l1[j]; a1[j + 8] = h1[j];
            }
            c0 = __builtin_amdgcn_wmma_f32_16x16x32_f16(false, a0, false, b, (short)0, c0, false, false);
            c1 = __builtin_amdgcn_wmma_f32_16x16x32_f16(false, a1, false, b, (short)0, c1, false, false);
        }
        const float bv = bias[col];
        const int rowOff = hi << 3;
#pragma unroll
        for (int r = 0; r < 8; ++r) {
            float v0 = c0[r] + bv, v1 = c1[r] + bv;
            if (relu) { v0 = fmaxf(v0, 0.0f); v1 = fmaxf(v1, 0.0f); }
            actOut[(size_t)(mBase + rowOff + r) * ldO + col]      = (half_t)v0;
            actOut[(size_t)(mBase + 16 + rowOff + r) * ldO + col] = (half_t)v1;
        }
    }
}

// Variant: epilogue stores f32 to global, no ReLU (dense_emb, nTiles==1).
__device__ __forceinline__ void gemm_layer_outf32(const half_t* __restrict__ actIn, int ldA,
                                                  const half_t* __restrict__ Wp, int KT,
                                                  const float*  __restrict__ bias,
                                                  float* __restrict__ gout, int ldO,
                                                  int nTiles, int tid) {
    const int lane = tid & 31, wave = tid >> 5;
    const int lm = lane & 15, hi = lane >> 4;
    const int units = nTiles * 2;
    for (int u = wave; u < units; u += NWAVES) {
        const int nT    = u >> 1;
        const int mBase = (u & 1) << 5;
        const int col   = (nT << 4) + lm;
        const half_t* bfrag = Wp + (((size_t)nT * KT) << 9) + (lane << 4);
        const half_t* ar0   = actIn + (size_t)(mBase + lm) * ldA + (hi << 3);
        const half_t* ar1   = ar0 + (size_t)16 * ldA;
        v8f c0 = {}, c1 = {};
        for (int kt = 0; kt < KT; ++kt) {
            v16h b = *(const v16h*)(bfrag + ((size_t)kt << 9));
            const half_t* p0 = ar0 + (kt << 5);
            const half_t* p1 = ar1 + (kt << 5);
            v8h l0 = *(const v8h*)p0, h0 = *(const v8h*)(p0 + 16);
            v8h l1 = *(const v8h*)p1, h1 = *(const v8h*)(p1 + 16);
            v16h a0, a1;
#pragma unroll
            for (int j = 0; j < 8; ++j) {
                a0[j] = l0[j]; a0[j + 8] = h0[j];
                a1[j] = l1[j]; a1[j + 8] = h1[j];
            }
            c0 = __builtin_amdgcn_wmma_f32_16x16x32_f16(false, a0, false, b, (short)0, c0, false, false);
            c1 = __builtin_amdgcn_wmma_f32_16x16x32_f16(false, a1, false, b, (short)0, c1, false, false);
        }
        const float bv = bias[col];
        const int rowOff = hi << 3;
#pragma unroll
        for (int r = 0; r < 8; ++r) {
            gout[(size_t)(mBase + rowOff + r) * ldO + col]      = c0[r] + bv;
            gout[(size_t)(mBase + 16 + rowOff + r) * ldO + col] = c1[r] + bv;
        }
    }
}

// ---------------------------------------------------------------------------
// Bottom MLP: dense[B][13] -> 512 -> 256 -> 64 -> 16, fused per 64-row block.
// ---------------------------------------------------------------------------
__global__ __launch_bounds__(256) void bot_kernel(
        const float* __restrict__ dense,
        const half_t* __restrict__ W0, const half_t* __restrict__ W1,
        const half_t* __restrict__ W2, const half_t* __restrict__ W3,
        const float* __restrict__ b0, const float* __restrict__ b1,
        const float* __restrict__ b2, const float* __restrict__ b3,
        float* __restrict__ demb) {
    __shared__ half_t sIn[ROWS * 32];    //   4 KB
    __shared__ half_t sA [ROWS * 512];   //  64 KB
    __shared__ half_t sB [ROWS * 256];   //  32 KB
    __shared__ half_t sC [ROWS * 64];    //   8 KB
    const int tid  = threadIdx.x;
    const int row0 = blockIdx.x * ROWS;
    for (int i = tid; i < ROWS * 32; i += 256) {
        int r = i >> 5, c = i & 31;
        float v = (c < 13) ? dense[(size_t)(row0 + r) * 13 + c] : 0.0f;
        sIn[i] = (half_t)v;
    }
    __syncthreads();
    gemm_layer(sIn, 32,  W0, 1,  b0, sA, 512, 32, true, tid);
    __syncthreads();
    gemm_layer(sA, 512,  W1, 16, b1, sB, 256, 16, true, tid);
    __syncthreads();
    gemm_layer(sB, 256,  W2, 8,  b2, sC, 64,  4,  true, tid);
    __syncthreads();
    gemm_layer_outf32(sC, 64, W3, 2, b3, demb + (size_t)row0 * 16, 16, 1, tid);
}

// ---------------------------------------------------------------------------
// Top MLP: gather 26 embeddings + dense_emb into LDS concat [64][448],
// then 448->256->128->64, per-thread dot + sigmoid for the final column.
// ---------------------------------------------------------------------------
__global__ __launch_bounds__(256) void top_kernel(
        const int* __restrict__ cat_idx, const float* __restrict__ emb,
        const float* __restrict__ demb,
        const half_t* __restrict__ T0, const half_t* __restrict__ T1,
        const half_t* __restrict__ T2,
        const float* __restrict__ tb0, const float* __restrict__ tb1,
        const float* __restrict__ tb2,
        const float* __restrict__ w3, const float* __restrict__ b3,
        float* __restrict__ out) {
    __shared__ half_t sCat[ROWS * 448];  //  56 KB
    __shared__ half_t sA  [ROWS * 256];  //  32 KB
    __shared__ half_t sB  [ROWS * 128];  //  16 KB
    __shared__ half_t sC  [ROWS * 64];   //   8 KB
    const int tid  = threadIdx.x;
    const int row0 = blockIdx.x * ROWS;
    // 28 segments of 16 halves per row: 26 gathers + dense_emb + zero pad
    for (int i = tid; i < ROWS * 28; i += 256) {
        int r = i / 28, s = i % 28;
        half_t* dst = &sCat[(size_t)r * 448 + s * 16];
        const float* src = nullptr;
        if (s < NUM_CAT) {
            int idx = cat_idx[(size_t)s * BATCH + row0 + r];
            src = emb + ((size_t)s * COUNT + (size_t)idx) * 16;
        } else if (s == NUM_CAT) {
            src = demb + (size_t)(row0 + r) * 16;
        }
        if (src) {
            const float4* s4 = (const float4*)src;
            float4 f0 = s4[0], f1 = s4[1], f2 = s4[2], f3 = s4[3];
            v8h h0 = {(half_t)f0.x, (half_t)f0.y, (half_t)f0.z, (half_t)f0.w,
                      (half_t)f1.x, (half_t)f1.y, (half_t)f1.z, (half_t)f1.w};
            v8h h1 = {(half_t)f2.x, (half_t)f2.y, (half_t)f2.z, (half_t)f2.w,
                      (half_t)f3.x, (half_t)f3.y, (half_t)f3.z, (half_t)f3.w};
            *(v8h*)dst = h0;
            *(v8h*)(dst + 8) = h1;
        } else {
            *(v8h*)dst = (v8h){};
            *(v8h*)(dst + 8) = (v8h){};
        }
    }
    __syncthreads();
    gemm_layer(sCat, 448, T0, 14, tb0, sA, 256, 16, true, tid);
    __syncthreads();
    gemm_layer(sA, 256,   T1, 8,  tb1, sB, 128, 8,  true, tid);
    __syncthreads();
    gemm_layer(sB, 128,   T2, 4,  tb2, sC, 64,  4,  true, tid);
    __syncthreads();
    if (tid < ROWS) {
        float acc = b3[0];
#pragma unroll 8
        for (int k = 0; k < 64; ++k) acc += (float)sC[(size_t)tid * 64 + k] * w3[k];
        out[row0 + tid] = 1.0f / (1.0f + expf(-acc));
    }
}

// ---------------------------------------------------------------------------
extern "C" void kernel_launch(void* const* d_in, const int* in_sizes, int n_in,
                              void* d_out, int out_size, void* d_ws, size_t ws_size,
                              hipStream_t stream) {
    const int*   cat_idx = (const int*)  d_in[0];
    const float* dense   = (const float*)d_in[1];
    const float* emb     = (const float*)d_in[2];
    const float* bw0 = (const float*)d_in[3];
    const float* bw1 = (const float*)d_in[4];
    const float* bw2 = (const float*)d_in[5];
    const float* bw3 = (const float*)d_in[6];
    const float* bb0 = (const float*)d_in[7];
    const float* bb1 = (const float*)d_in[8];
    const float* bb2 = (const float*)d_in[9];
    const float* bb3 = (const float*)d_in[10];
    const float* tw0 = (const float*)d_in[11];
    const float* tw1 = (const float*)d_in[12];
    const float* tw2 = (const float*)d_in[13];
    const float* tw3 = (const float*)d_in[14];
    const float* tb0 = (const float*)d_in[15];
    const float* tb1 = (const float*)d_in[16];
    const float* tb2 = (const float*)d_in[17];
    const float* tb3 = (const float*)d_in[18];
    float* out = (float*)d_out;

    // Workspace layout
    char* ws = (char*)d_ws;
    size_t off = 0;
    auto take = [&](size_t bytes) { char* p = ws + off; off += (bytes + 255) & ~(size_t)255; return p; };
    float*  demb = (float*) take((size_t)BATCH * 16 * sizeof(float));
    half_t* W0p  = (half_t*)take((size_t)1  * 32 * 512 * 2);
    half_t* W1p  = (half_t*)take((size_t)16 * 16 * 512 * 2);
    half_t* W2p  = (half_t*)take((size_t)8  * 4  * 512 * 2);
    half_t* W3p  = (half_t*)take((size_t)2  * 1  * 512 * 2);
    half_t* T0p  = (half_t*)take((size_t)14 * 16 * 512 * 2);
    half_t* T1p  = (half_t*)take((size_t)8  * 8  * 512 * 2);
    half_t* T2p  = (half_t*)take((size_t)4  * 4  * 512 * 2);
    (void)ws_size; (void)in_sizes; (void)n_in; (void)out_size;

    auto pack = [&](const float* s, half_t* d, int K, int N, int KT, int NT) {
        int n = KT * NT * 512;
        pack_b_kernel<<<(n + 255) / 256, 256, 0, stream>>>(s, d, K, N, KT, NT);
    };
    pack(bw0, W0p, 13,  512, 1,  32);
    pack(bw1, W1p, 512, 256, 16, 16);
    pack(bw2, W2p, 256, 64,  8,  4);
    pack(bw3, W3p, 64,  16,  2,  1);
    pack(tw0, T0p, 432, 256, 14, 16);
    pack(tw1, T1p, 256, 128, 8,  8);
    pack(tw2, T2p, 128, 64,  4,  4);

    bot_kernel<<<BATCH / ROWS, 256, 0, stream>>>(dense, W0p, W1p, W2p, W3p,
                                                 bb0, bb1, bb2, bb3, demb);
    top_kernel<<<BATCH / ROWS, 256, 0, stream>>>(cat_idx, emb, demb,
                                                 T0p, T1p, T2p,
                                                 tb0, tb1, tb2,
                                                 tw3, tb3, out);
}
